// NeuralMemory_40527311405389
// MI455X (gfx1250) — matrix operand order repaired and checked by reading, not compiled, not run on hardware
//
#include <hip/hip_runtime.h>

#define BB 2
#define SEQ 4096
#define DIM 1024
#define HEADS 4
#define DHD 256
#define DI 1024
#define CHUNK 64
#define NCHUNK (SEQ / CHUNK)   // 64
#define BH (BB * HEADS)        // 8
#define DEPTH 2
#define MAX_LR 1e-2f
#define EPSF 1e-6f

typedef __attribute__((ext_vector_type(16))) _Float16 v16h;
typedef __attribute__((ext_vector_type(8)))  _Float16 v8h;
typedef __attribute__((ext_vector_type(8)))  float    v8f;

__device__ __forceinline__ v8f wmma_f16(v16h a, v16h b, v8f c) {
  return __builtin_amdgcn_wmma_f32_16x16x32_f16(false, a, false, b, (short)0, c,
                                                false, false);
}

__device__ __forceinline__ v16h cat16(v8h lo, v8h hi) {
  return __builtin_shufflevector(lo, hi, 0, 1, 2, 3, 4, 5, 6, 7,
                                 8, 9, 10, 11, 12, 13, 14, 15);
}

// A fragment 16x32 (MxK), row-major source: element (m,k) at p[m*ld + k].
// Lane layout (ISA 7.12.2): two contiguous 8-half runs -> two 16B loads.
__device__ __forceinline__ v16h load_a(const _Float16* p, int ld) {
  int lane = threadIdx.x & 31;
  const _Float16* q = p + (lane & 15) * ld + ((lane >> 4) << 3);
  return cat16(*(const v8h*)q, *(const v8h*)(q + 16));
}
// B fragment 32x16 (KxN) from N-major storage: element (k,n) at p[n*ld + k].
// Lane layout matches load_a: two contiguous 8-half K-runs -> two 16B loads.
__device__ __forceinline__ v16h load_bT(const _Float16* p, int ld) {
  int lane = threadIdx.x & 31;
  const _Float16* q = p + (lane & 15) * ld + ((lane >> 4) << 3);
  return cat16(*(const v8h*)q, *(const v8h*)(q + 16));
}

__device__ __forceinline__ float sigmoidf_(float x) { return 1.f / (1.f + __expf(-x)); }

// ---------------- weight converts ----------------
__global__ void cvt_kernel(const float* __restrict__ src, _Float16* __restrict__ dst, int n) {
  int i = blockIdx.x * 256 + threadIdx.x;
  if (i < n) dst[i] = (_Float16)src[i];
}
// mem_weights (DEPTH,DH,DH) -> f16 transposed per depth: dst[d][n][k] = src[d][k][n]
__global__ void cvtT_mem_kernel(const float* __restrict__ src, _Float16* __restrict__ dst) {
  int idx = blockIdx.x * 256 + threadIdx.x;
  if (idx >= DEPTH * DHD * DHD) return;
  int d = idx >> 16, r = idx & 65535;
  int k = r >> 8, n = r & 255;
  dst[(d << 16) + (n << 8) + k] = (_Float16)src[idx];
}

// ---------------- rmsnorm (store + shifted retrieve) + per-token gates ----------------
__global__ void prep_kernel(const float* __restrict__ seq,
                            const float* __restrict__ store_w, const float* __restrict__ ret_w,
                            const float* __restrict__ Wstep, const float* __restrict__ Wgate,
                            _Float16* __restrict__ s16, _Float16* __restrict__ r16,
                            float* __restrict__ lr, float* __restrict__ gate) {
  int t = blockIdx.x;               // 0 .. B*SEQ-1
  int b = t / SEQ, i = t % SEQ;
  int tid = threadIdx.x;
  __shared__ float red[256];
  __shared__ float row[DIM];

  const float* x = seq + (size_t)t * DIM;
  float ss = 0.f;
  for (int d = tid; d < DIM; d += 256) { float v = x[d]; row[d] = v; ss += v * v; }
  red[tid] = ss; __syncthreads();
  for (int off = 128; off > 0; off >>= 1) { if (tid < off) red[tid] += red[tid + off]; __syncthreads(); }
  float rinv = rsqrtf(red[0] / DIM + EPSF);
  __syncthreads();
  float acc[HEADS] = {0.f, 0.f, 0.f, 0.f};
  for (int d = tid; d < DIM; d += 256) {
    float sv = row[d] * rinv * store_w[d];
    s16[(size_t)t * DIM + d] = (_Float16)sv;
#pragma unroll
    for (int h = 0; h < HEADS; ++h) acc[h] += sv * Wstep[h * DIM + d];
  }
#pragma unroll
  for (int h = 0; h < HEADS; ++h) {
    red[tid] = acc[h]; __syncthreads();
    for (int off = 128; off > 0; off >>= 1) { if (tid < off) red[tid] += red[tid + off]; __syncthreads(); }
    if (tid == 0) {
      int ch = i / CHUNK, c = i % CHUNK;
      lr[(((size_t)b * HEADS + h) * NCHUNK + ch) * CHUNK + c] = MAX_LR * sigmoidf_(red[0]);
    }
    __syncthreads();
  }

  int src = i + (CHUNK - 1);
  float racc[HEADS] = {0.f, 0.f, 0.f, 0.f};
  if (src < SEQ) {
    const float* xr = seq + ((size_t)b * SEQ + src) * DIM;
    float rs = 0.f;
    for (int d = tid; d < DIM; d += 256) { float v = xr[d]; row[d] = v; rs += v * v; }
    red[tid] = rs; __syncthreads();
    for (int off = 128; off > 0; off >>= 1) { if (tid < off) red[tid] += red[tid + off]; __syncthreads(); }
    float rinv2 = rsqrtf(red[0] / DIM + EPSF);
    __syncthreads();
    for (int d = tid; d < DIM; d += 256) {
      float rv = row[d] * rinv2 * ret_w[d];
      r16[(size_t)t * DIM + d] = (_Float16)rv;
#pragma unroll
      for (int h = 0; h < HEADS; ++h) racc[h] += rv * Wgate[h * DIM + d];
    }
  } else {
    for (int d = tid; d < DIM; d += 256) r16[(size_t)t * DIM + d] = (_Float16)0.f;
  }
#pragma unroll
  for (int h = 0; h < HEADS; ++h) {
    red[tid] = racc[h]; __syncthreads();
    for (int off = 128; off > 0; off >>= 1) { if (tid < off) red[tid] += red[tid + off]; __syncthreads(); }
    if (tid == 0) gate[((size_t)b * HEADS + h) * SEQ + i] = sigmoidf_(red[0]);
    __syncthreads();
  }
}

// ---------------- chunk-mean -> momentum / decay gates ----------------
__global__ void chunkstat_kernel(const _Float16* __restrict__ s16,
                                 const float* __restrict__ Wmom, const float* __restrict__ Wdec,
                                 float* __restrict__ mom, float* __restrict__ dec) {
  int bc = blockIdx.x;
  int b = bc / NCHUNK, ch = bc % NCHUNK;
  int tid = threadIdx.x;
  __shared__ float mean[DIM];
  __shared__ float red[256];
  for (int d = tid; d < DIM; d += 256) {
    float a = 0.f;
    for (int c = 0; c < CHUNK; ++c)
      a += (float)s16[((size_t)b * SEQ + ch * CHUNK + c) * DIM + d];
    mean[d] = a * (1.f / CHUNK);
  }
  __syncthreads();
#pragma unroll
  for (int h = 0; h < HEADS; ++h) {
    float am = 0.f, ad = 0.f;
    for (int d = tid; d < DIM; d += 256) { am += mean[d] * Wmom[h * DIM + d]; ad += mean[d] * Wdec[h * DIM + d]; }
    red[tid] = am; __syncthreads();
    for (int off = 128; off > 0; off >>= 1) { if (tid < off) red[tid] += red[tid + off]; __syncthreads(); }
    float mres = red[0]; __syncthreads();
    red[tid] = ad; __syncthreads();
    for (int off = 128; off > 0; off >>= 1) { if (tid < off) red[tid] += red[tid + off]; __syncthreads(); }
    if (tid == 0) {
      mom[((size_t)b * HEADS + h) * NCHUNK + ch] = sigmoidf_(mres);
      dec[((size_t)b * HEADS + h) * NCHUNK + ch] = sigmoidf_(red[0]);
    }
    __syncthreads();
  }
}

// ---------------- WMMA projection: Y = X(8192xDIM) @ W(ExDIM)^T, head-split scatter ----------------
#define LDT 40   // padded K-extent (halves), multiple of 8 -> 16B alignment preserved
__device__ __forceinline__ void scatter_headsplit(v8f c, int m0, int n0, int mt, int nt,
                                                  _Float16* dstK, _Float16* dstV) {
  int lane = threadIdx.x & 31;
  int nl = lane & 15, mb = (lane >> 4) * 8;
#pragma unroll
  for (int i = 0; i < 8; ++i) {
    int t = m0 + mt * 16 + mb + i;
    int e = n0 + nt * 16 + nl;
    int b = t / SEQ, ii = t % SEQ;
    int ch = ii / CHUNK, cc = ii % CHUNK;
    _Float16* dst = dstK;
    if (e >= DI) { dst = dstV; e -= DI; }
    int h = e >> 8, dh = e & 255;
    dst[((((size_t)b * HEADS + h) * NCHUNK + ch) * CHUNK + cc) * DHD + dh] = (_Float16)c[i];
  }
}

__global__ void proj_kernel(const _Float16* __restrict__ X, const _Float16* __restrict__ W,
                            _Float16* __restrict__ dstK, _Float16* __restrict__ dstV) {
  __shared__ alignas(16) _Float16 At[64 * LDT];   // 64 rows x 32 k (row-major)
  __shared__ alignas(16) _Float16 Bt[64 * LDT];   // 64 n-rows x 32 k (N-major == W rows)
  int tid = threadIdx.x, wave = tid >> 5;
  int m0 = blockIdx.x * 64, n0 = blockIdx.y * 64;
  int mt0 = (wave * 2) >> 2, nt0 = (wave * 2) & 3;
  int mt1 = (wave * 2 + 1) >> 2, nt1 = (wave * 2 + 1) & 3;
  int r = tid >> 2, ko = (tid & 3) * 8;
  v8f acc0 = {}, acc1 = {};
  for (int k0 = 0; k0 < DIM; k0 += 32) {
    __syncthreads();
    *(v8h*)&At[r * LDT + ko] = *(const v8h*)&X[(size_t)(m0 + r) * DIM + k0 + ko];
    *(v8h*)&Bt[r * LDT + ko] = *(const v8h*)&W[(size_t)(n0 + r) * DIM + k0 + ko];
    if (k0 + 32 < DIM)
      __builtin_prefetch(&X[(size_t)(m0 + r) * DIM + k0 + 32 + ko], 0, 0);
    __syncthreads();
    acc0 = wmma_f16(load_a(At + mt0 * 16 * LDT, LDT), load_bT(Bt + nt0 * 16 * LDT, LDT), acc0);
    acc1 = wmma_f16(load_a(At + mt1 * 16 * LDT, LDT), load_bT(Bt + nt1 * 16 * LDT, LDT), acc1);
  }
  scatter_headsplit(acc0, m0, n0, mt0, nt0, dstK, dstV);
  scatter_headsplit(acc1, m0, n0, mt1, nt1, dstK, dstV);
}

// ---------------- fused per-chunk gradient: surprise^T = -grad^T (5 WMMA GEMMs) ----------------
// surprise stored TRANSPOSED per depth: surT[d][q][p] = -grad[d][p][q]  (lane-contiguous stores)
__global__ void grad_kernel(const _Float16* __restrict__ k16, const _Float16* __restrict__ v16,
                            const float* __restrict__ lr, const _Float16* __restrict__ mem16,
                            const _Float16* __restrict__ memT16,
                            float* __restrict__ surprise) {
  __shared__ alignas(16) _Float16 Kc[CHUNK * DHD];    // k chunk, row-major (token x dim)
  __shared__ alignas(16) _Float16 KcT[DHD * CHUNK];   // k^T       (dim x token)
  __shared__ alignas(16) _Float16 Vc[CHUNK * DHD];    // v -> dpred (token x dh)
  __shared__ alignas(16) _Float16 DPT[DHD * CHUNK];   // dpred^T   (dh x token)
  __shared__ alignas(16) _Float16 S0[CHUNK * DHD];    // silu(h0) -> reused as dh0^T
  __shared__ alignas(16) _Float16 S0T[DHD * CHUNK];   // silu(h0)^T
  __shared__ alignas(16) _Float16 DS[CHUNK * DHD];    // silu'(h0) (token x dh)
  __shared__ float lrs[CHUNK];
  int tid = threadIdx.x, wave = tid >> 5, lane = tid & 31;
  int nl = lane & 15, mb = (lane >> 4) * 8;
  int cidx = blockIdx.x;
  size_t base = (size_t)cidx * (CHUNK * DHD);

  for (int c8 = tid; c8 < (CHUNK * DHD) / 8; c8 += 256) {
    int m = c8 >> 5, n0 = (c8 & 31) * 8;
    v8h kv = *(const v8h*)(k16 + base + (size_t)c8 * 8);
    v8h vv = *(const v8h*)(v16 + base + (size_t)c8 * 8);
    *(v8h*)&Kc[c8 * 8] = kv;
    *(v8h*)&Vc[c8 * 8] = vv;
#pragma unroll
    for (int j = 0; j < 8; ++j) KcT[(n0 + j) * CHUNK + m] = kv[j];
  }
  if (tid < CHUNK) lrs[tid] = lr[(size_t)cidx * CHUNK + tid];
  __syncthreads();

  // GEMM1: h0 = K @ W0 (A = Kc row-major, B = W0 via W0T N-major global)
  for (int tq = wave; tq < 64; tq += 8) {
    int mt = tq >> 4, nt = tq & 15;
    v8f c = {};
    for (int kk = 0; kk < DHD; kk += 32)
      c = wmma_f16(load_a(Kc + mt * 16 * DHD + kk, DHD),
                   load_bT(memT16 + (size_t)(nt * 16) * DHD + kk, DHD), c);
#pragma unroll
    for (int i = 0; i < 8; ++i) {
      int m = mt * 16 + mb + i, n = nt * 16 + nl;
      float h = c[i], sg = sigmoidf_(h);
      S0[m * DHD + n]  = (_Float16)(h * sg);
      S0T[n * CHUNK + m] = (_Float16)(h * sg);
      DS[m * DHD + n]  = (_Float16)(sg * (1.f + h * (1.f - sg)));
    }
  }
  __syncthreads();

  // GEMM2: pred = S0 @ W1 (B = W1 via W1T) -> dpred into Vc (in place) + DPT
  for (int tq = wave; tq < 64; tq += 8) {
    int mt = tq >> 4, nt = tq & 15;
    v8f c = {};
    for (int kk = 0; kk < DHD; kk += 32)
      c = wmma_f16(load_a(S0 + mt * 16 * DHD + kk, DHD),
                   load_bT(memT16 + (size_t)DHD * DHD + (size_t)(nt * 16) * DHD + kk, DHD), c);
#pragma unroll
    for (int i = 0; i < 8; ++i) {
      int m = mt * 16 + mb + i, n = nt * 16 + nl;
      float d = (2.0f / DHD) * lrs[m] * (c[i] - (float)Vc[m * DHD + n]);
      Vc[m * DHD + n]   = (_Float16)d;
      DPT[n * CHUNK + m] = (_Float16)d;
    }
  }
  __syncthreads();

  float* sur = surprise + (size_t)cidx * (DEPTH * DHD * DHD);
  _Float16* DhT = S0;   // S0 dead after GEMM2; reuse as dh0^T (dh x token)

  // GEMM3: gW1^T = DP^T @ S0  (A = DPT, B = S0 via S0T; K = 64) -> surT[1] coalesced
  for (int tq = wave; tq < 256; tq += 8) {
    int mt = tq >> 4, nt = tq & 15;
    v8f c = {};
    for (int kk = 0; kk < CHUNK; kk += 32)
      c = wmma_f16(load_a(DPT + mt * 16 * CHUNK + kk, CHUNK),
                   load_bT(S0T + (nt * 16) * CHUNK + kk, CHUNK), c);
#pragma unroll
    for (int i = 0; i < 8; ++i) {
      int m = mt * 16 + mb + i, n = nt * 16 + nl;
      sur[(size_t)DHD * DHD + m * DHD + n] = -c[i];
    }
  }

  // GEMM4: dh0^T = W1 @ DP^T, times silu'(h0)  (A = W1 row-major global, B = DP^T via Vc)
  for (int tq = wave; tq < 64; tq += 8) {
    int mt = tq >> 2, nt = tq & 3;     // M=256 (dh), N=64 (token)
    v8f c = {};
    for (int kk = 0; kk < DHD; kk += 32)
      c = wmma_f16(load_a(mem16 + (size_t)DHD * DHD + mt * 16 * DHD + kk, DHD),
                   load_bT(Vc + (size_t)(nt * 16) * DHD + kk, DHD), c);
#pragma unroll
    for (int i = 0; i < 8; ++i) {
      int m = mt * 16 + mb + i, n = nt * 16 + nl;     // m = dh, n = token
      DhT[m * CHUNK + n] = (_Float16)(c[i] * (float)DS[n * DHD + m]);
    }
  }
  __syncthreads();

  // GEMM5: gW0^T = dh0^T @ K  (A = DhT, B = K via KcT; K = 64) -> surT[0] coalesced
  for (int tq = wave; tq < 256; tq += 8) {
    int mt = tq >> 4, nt = tq & 15;
    v8f c = {};
    for (int kk = 0; kk < CHUNK; kk += 32)
      c = wmma_f16(load_a(DhT + mt * 16 * CHUNK + kk, CHUNK),
                   load_bT(KcT + (nt * 16) * CHUNK + kk, CHUNK), c);
#pragma unroll
    for (int i = 0; i < 8; ++i) {
      int m = mt * 16 + mb + i, n = nt * 16 + nl;
      sur[m * DHD + n] = -c[i];
    }
  }
}

// ---------------- gated scans (transposed domain); emit f16 per-chunk W^T ----------------
// element e = (d, q, p); Wupd[d][q][p] = mem[d][p][q] + updates[d][p][q]
__global__ void scan_kernel(const float* __restrict__ surprise,
                            const float* __restrict__ mom, const float* __restrict__ dec,
                            const float* __restrict__ memw, _Float16* __restrict__ Wupd) {
  int e = blockIdx.x * 256 + threadIdx.x;
  int bhi = blockIdx.y;
  const size_t stride = (size_t)DEPTH * DHD * DHD;
  int d = e >> 16, rr = e & 65535;
  int q = rr >> 8, p = rr & 255;
  float basev = memw[(d << 16) + (p << 8) + q];
  float m = 0.f, u = 0.f;
  for (int t = 0; t < NCHUNK; ++t) {
    float s = surprise[((size_t)bhi * NCHUNK + t) * stride + e];
    m = mom[bhi * NCHUNK + t] * m + s;
    u = (1.f - dec[bhi * NCHUNK + t]) * u + m;
    Wupd[((size_t)bhi * NCHUNK + t) * stride + e] = (_Float16)(basev + u);
  }
}

// ---------------- retrieve: q @ W0, silu, @ W1, multihead rmsnorm * (gamma+1) * gate ----------------
__global__ void retrieve_kernel(const _Float16* __restrict__ q16, const _Float16* __restrict__ Wupd,
                                const float* __restrict__ gamma, const float* __restrict__ gate,
                                _Float16* __restrict__ vals16) {
  __shared__ alignas(16) _Float16 Q[CHUNK * DHD];
  __shared__ alignas(16) _Float16 X1[CHUNK * DHD];
  __shared__ float Y[CHUNK * DHD];
  __shared__ float rowinv[CHUNK];
  int tid = threadIdx.x, wave = tid >> 5, lane = tid & 31;
  int nl = lane & 15, mb = (lane >> 4) * 8;
  int cidx = blockIdx.x;
  int bhi = cidx / NCHUNK, ch = cidx % NCHUNK;
  int b = bhi / HEADS, h = bhi % HEADS;
  size_t base = (size_t)cidx * (CHUNK * DHD);
  for (int c8 = tid; c8 < (CHUNK * DHD) / 8; c8 += 256)
    *(v8h*)&Q[c8 * 8] = *(const v8h*)(q16 + base + (size_t)c8 * 8);
  __syncthreads();
  const _Float16* W0 = Wupd + (size_t)cidx * DEPTH * DHD * DHD;   // stored N-major (q x p)
  const _Float16* W1 = W0 + DHD * DHD;
  for (int tq = wave; tq < 64; tq += 8) {
    int mt = tq >> 4, nt = tq & 15;
    v8f c = {};
    for (int kk = 0; kk < DHD; kk += 32)
      c = wmma_f16(load_a(Q + mt * 16 * DHD + kk, DHD),
                   load_bT(W0 + (size_t)(nt * 16) * DHD + kk, DHD), c);
#pragma unroll
    for (int i = 0; i < 8; ++i) {
      int m = mt * 16 + mb + i, n = nt * 16 + nl;
      float x = c[i];
      X1[m * DHD + n] = (_Float16)(x * sigmoidf_(x));
    }
  }
  __syncthreads();
  for (int tq = wave; tq < 64; tq += 8) {
    int mt = tq >> 4, nt = tq & 15;
    v8f c = {};
    for (int kk = 0; kk < DHD; kk += 32)
      c = wmma_f16(load_a(X1 + mt * 16 * DHD + kk, DHD),
                   load_bT(W1 + (size_t)(nt * 16) * DHD + kk, DHD), c);
#pragma unroll
    for (int i = 0; i < 8; ++i) {
      int m = mt * 16 + mb + i, n = nt * 16 + nl;
      Y[m * DHD + n] = c[i];
    }
  }
  __syncthreads();
  if (tid < CHUNK) {
    float ss = 0.f;
    for (int n = 0; n < DHD; ++n) { float y = Y[tid * DHD + n]; ss += y * y; }
    rowinv[tid] = rsqrtf(ss / DHD + EPSF);
  }
  __syncthreads();
  for (int c8 = tid; c8 < (CHUNK * DHD) / 8; c8 += 256) {
    int m = c8 >> 5, n0 = (c8 & 31) * 8;
    int token = ch * CHUNK + m;
    float g = gate[((size_t)b * HEADS + h) * SEQ + token] * rowinv[m];
    v8h o;
#pragma unroll
    for (int j = 0; j < 8; ++j)
      o[j] = (_Float16)(Y[m * DHD + n0 + j] * g * (gamma[h * DHD + n0 + j] + 1.f));
    *(v8h*)&vals16[((size_t)b * SEQ + token) * DI + h * DHD + n0] = o;
  }
}

// ---------------- output projection with +63 shift ----------------
__global__ void comb_kernel(const _Float16* __restrict__ V, const _Float16* __restrict__ Wc,
                            float* __restrict__ out) {
  __shared__ alignas(16) _Float16 At[64 * LDT];
  __shared__ alignas(16) _Float16 Bt[64 * LDT];
  int tid = threadIdx.x, wave = tid >> 5, lane = tid & 31;
  int nl = lane & 15, mb = (lane >> 4) * 8;
  int m0 = blockIdx.x * 64, n0 = blockIdx.y * 64;
  int mt0 = (wave * 2) >> 2, nt0 = (wave * 2) & 3;
  int mt1 = (wave * 2 + 1) >> 2, nt1 = (wave * 2 + 1) & 3;
  int r = tid >> 2, ko = (tid & 3) * 8;
  v8f acc0 = {}, acc1 = {};
  for (int k0 = 0; k0 < DI; k0 += 32) {
    __syncthreads();
    *(v8h*)&At[r * LDT + ko] = *(const v8h*)&V[(size_t)(m0 + r) * DI + k0 + ko];
    *(v8h*)&Bt[r * LDT + ko] = *(const v8h*)&Wc[(size_t)(n0 + r) * DI + k0 + ko];
    if (k0 + 32 < DI)
      __builtin_prefetch(&V[(size_t)(m0 + r) * DI + k0 + 32 + ko], 0, 0);
    __syncthreads();
    acc0 = wmma_f16(load_a(At + mt0 * 16 * LDT, LDT), load_bT(Bt + nt0 * 16 * LDT, LDT), acc0);
    acc1 = wmma_f16(load_a(At + mt1 * 16 * LDT, LDT), load_bT(Bt + nt1 * 16 * LDT, LDT), acc1);
  }
#pragma unroll
  for (int half = 0; half < 2; ++half) {
    int mt = half ? mt1 : mt0, nt = half ? nt1 : nt0;
    v8f c = half ? acc1 : acc0;
#pragma unroll
    for (int i = 0; i < 8; ++i) {
      int t = m0 + mt * 16 + mb + i;
      int d = n0 + nt * 16 + nl;
      int b = t / SEQ, ii = t % SEQ;
      int opos = ii + (CHUNK - 1);
      if (opos < SEQ) out[((size_t)b * SEQ + opos) * DIM + d] = c[i];
    }
  }
}

__global__ void fill_kernel(const float* __restrict__ emb, float* __restrict__ out) {
  int idx = blockIdx.x * 256 + threadIdx.x;
  if (idx >= BB * (CHUNK - 1) * DIM) return;
  int d = idx % DIM;
  int r = idx / DIM;
  int b = r / (CHUNK - 1);
  int i = r % (CHUNK - 1);
  out[((size_t)b * SEQ + i) * DIM + d] = emb[d];
}

extern "C" void kernel_launch(void* const* d_in, const int* in_sizes, int n_in,
                              void* d_out, int out_size, void* d_ws, size_t ws_size,
                              hipStream_t stream) {
  (void)in_sizes; (void)n_in; (void)out_size; (void)ws_size;
  const float* seq   = (const float*)d_in[0];
  const float* snw   = (const float*)d_in[1];
  const float* rnw   = (const float*)d_in[2];
  const float* gamma = (const float*)d_in[3];
  const float* memw  = (const float*)d_in[4];
  const float* Wq    = (const float*)d_in[5];
  const float* Wkv   = (const float*)d_in[6];
  const float* Wstep = (const float*)d_in[7];
  const float* Wmom  = (const float*)d_in[8];
  const float* Wdec  = (const float*)d_in[9];
  const float* Wgate = (const float*)d_in[10];
  const float* Wcomb = (const float*)d_in[11];
  const float* emb   = (const float*)d_in[12];
  float* out = (float*)d_out;

  char* p = (char*)d_ws;
  auto carve = [&](size_t bytes) -> char* {
    char* r = p;
    p += (bytes + 255) & ~(size_t)255;
    return r;
  };
  _Float16* Wq16    = (_Float16*)carve((size_t)DI * DIM * 2);
  _Float16* Wkv16   = (_Float16*)carve((size_t)2 * DI * DIM * 2);
  _Float16* Wcomb16 = (_Float16*)carve((size_t)DIM * DI * 2);
  _Float16* mem16   = (_Float16*)carve((size_t)DEPTH * DHD * DHD * 2);
  _Float16* memT16  = (_Float16*)carve((size_t)DEPTH * DHD * DHD * 2);
  _Float16* s16     = (_Float16*)carve((size_t)BB * SEQ * DIM * 2);
  _Float16* r16     = (_Float16*)carve((size_t)BB * SEQ * DIM * 2);
  _Float16* q16     = (_Float16*)carve((size_t)BB * SEQ * DI * 2);
  _Float16* k16     = (_Float16*)carve((size_t)BB * SEQ * DI * 2);
  _Float16* v16     = (_Float16*)carve((size_t)BB * SEQ * DI * 2);
  float* lr         = (float*)carve((size_t)BH * SEQ * 4);
  float* gatep      = (float*)carve((size_t)BH * SEQ * 4);
  float* mom        = (float*)carve((size_t)BH * NCHUNK * 4);
  float* dec        = (float*)carve((size_t)BH * NCHUNK * 4);
  float* surprise   = (float*)carve((size_t)BH * NCHUNK * DEPTH * DHD * DHD * 4);
  _Float16* Wupd    = (_Float16*)carve((size_t)BH * NCHUNK * DEPTH * DHD * DHD * 2);
  _Float16* vals16  = (_Float16*)carve((size_t)BB * SEQ * DI * 2);

  cvt_kernel<<<(DI * DIM + 255) / 256, 256, 0, stream>>>(Wq, Wq16, DI * DIM);
  cvt_kernel<<<(2 * DI * DIM + 255) / 256, 256, 0, stream>>>(Wkv, Wkv16, 2 * DI * DIM);
  cvt_kernel<<<(DIM * DI + 255) / 256, 256, 0, stream>>>(Wcomb, Wcomb16, DIM * DI);
  cvt_kernel<<<(DEPTH * DHD * DHD + 255) / 256, 256, 0, stream>>>(memw, mem16, DEPTH * DHD * DHD);
  cvtT_mem_kernel<<<(DEPTH * DHD * DHD + 255) / 256, 256, 0, stream>>>(memw, memT16);

  prep_kernel<<<BB * SEQ, 256, 0, stream>>>(seq, snw, rnw, Wstep, Wgate, s16, r16, lr, gatep);
  chunkstat_kernel<<<BB * NCHUNK, 256, 0, stream>>>(s16, Wmom, Wdec, mom, dec);

  proj_kernel<<<dim3(BB * SEQ / 64, DI / 64), 256, 0, stream>>>(r16, Wq16, q16, nullptr);
  proj_kernel<<<dim3(BB * SEQ / 64, 2 * DI / 64), 256, 0, stream>>>(s16, Wkv16, k16, v16);

  grad_kernel<<<BH * NCHUNK, 256, 0, stream>>>(k16, v16, lr, mem16, memT16, surprise);
  scan_kernel<<<dim3(DEPTH * DHD * DHD / 256, BH), 256, 0, stream>>>(surprise, mom, dec, memw, Wupd);
  retrieve_kernel<<<BH * NCHUNK, 256, 0, stream>>>(q16, Wupd, gamma, gatep, vals16);

  comb_kernel<<<dim3(BB * SEQ / 64, DIM / 64), 256, 0, stream>>>(vals16, Wcomb16, out);
  fill_kernel<<<(BB * (CHUNK - 1) * DIM + 255) / 256, 256, 0, stream>>>(emb, out);
}
